// Dynamic_class_58935541236107
// MI455X (gfx1250) — matrix operand order repaired and checked by reading, not compiled, not run on hardware
//
#include <hip/hip_runtime.h>
#include <hip/hip_bf16.h>
#include <math.h>

typedef _Float16 half_t;
typedef __attribute__((ext_vector_type(16))) _Float16 v16h;
typedef __attribute__((ext_vector_type(8)))  _Float16 v8h;
typedef __attribute__((ext_vector_type(8)))  float    v8f;

#define NNODES 32768
#define NB     64
#define NPTS   512
#define NE     524288
#define KNN    16

#define GF_ACCUM 1
#define GF_BIAS  2
#define GF_RELU  4

// ---------------------------------------------------------------- utilities
__global__ void zero_f32(float* __restrict__ p, long n) {
  long i = (long)blockIdx.x * blockDim.x + threadIdx.x;
  long stride = (long)gridDim.x * blockDim.x;
  for (; i < n; i += stride) p[i] = 0.f;
}

__global__ void deg_kernel(const int* __restrict__ dst, float* __restrict__ deg, int E) {
  int e = blockIdx.x * blockDim.x + threadIdx.x;
  if (e < E) atomicAdd(&deg[dst[e]], 1.0f);
}

__global__ void norm_kernel(const int* __restrict__ src, const int* __restrict__ dst,
                            const float* __restrict__ deg, float* __restrict__ nrm, int E) {
  int e = blockIdx.x * blockDim.x + threadIdx.x;
  if (e < E) {
    float ds = deg[src[e]], dd = deg[dst[e]];
    float a = ds > 0.f ? rsqrtf(fmaxf(ds, 1.f)) : 0.f;
    float b = dd > 0.f ? rsqrtf(fmaxf(dd, 1.f)) : 0.f;
    nrm[e] = a * b;
  }
}

// hn[dst,f] += norm[e] * h[src,f];  edge = x-dim thread, feature = blockIdx.y
__global__ void hop_kernel(const float* __restrict__ h, const int* __restrict__ src,
                           const int* __restrict__ dst, const float* __restrict__ nrm,
                           float* __restrict__ hn, int F) {
  int e = blockIdx.x * blockDim.x + threadIdx.x;
  int f = blockIdx.y;
  if (e < NE) {
    atomicAdd(&hn[(long)dst[e] * F + f], nrm[e] * h[(long)src[e] * F + f]);
  }
}

// ------------------------------------------------ WMMA GEMM: C = A(MxK)@W(KxN)
// W staged per-block into LDS: zero-padded, column-major with +8-half skew so
// B fragments are two ds_load_b128 per K-step and bank-conflict-free.
// FLAGS is compile-time so the epilogue is branch-free straight-line code.
template <int K, int N, int FLAGS>
__global__ __launch_bounds__(128) void gemm_wmma(
    const float* __restrict__ A, const float* __restrict__ W,
    const float* __restrict__ bias, float* __restrict__ C, int M)
{
  constexpr int KP  = (K + 31) & ~31;    // padded K
  constexpr int NPD = (N + 15) & ~15;    // padded N
  constexpr int WS  = KP + 8;            // skewed col stride (halves)
  extern __shared__ half_t Wl[];         // NPD * WS halves, col-major

  for (int t = threadIdx.x; t < NPD * KP; t += 128) {
    int n = t / KP, k = t - n * KP;
    Wl[n * WS + k] = (k < K && n < N) ? (half_t)W[k * N + n] : (half_t)0.f;
  }
  __syncthreads();

  const int lane = threadIdx.x & 31;
  const int wave = threadIdx.x >> 5;
  constexpr int tilesN = NPD >> 4;
  const int tilesTotal = (M >> 4) * tilesN;
  int tile = blockIdx.x * 4 + wave;
  if (tile >= tilesTotal) return;        // uniform per wave (after the barrier)

  const int tm  = tile / tilesN;
  const int tn  = tile - tm * tilesN;
  const int l15 = lane & 15;
  const int hi  = lane >> 4;
  const int row = tm * 16 + l15;
  const int col = tn * 16 + l15;
  const int kbA = hi ? 8 : 0;            // A frag: low lanes K 0..7/16..23

  const bool colOk = (N == NPD) || (col < N);
  float* Ccol = C + (long)(tm * 16 + (hi ? 8 : 0)) * N + col;

  v8f acc;
  if constexpr (FLAGS & GF_ACCUM) {
#pragma unroll
    for (int r = 0; r < 8; ++r) acc[r] = colOk ? Ccol[(long)r * N] : 0.f;
  } else {
#pragma unroll
    for (int r = 0; r < 8; ++r) acc[r] = 0.f;
  }

  const float*  Arow = A + (long)row * K;
  const half_t* wcol = Wl + col * WS + (hi ? 8 : 0);    // B frag col base

#pragma unroll
  for (int kk = 0; kk < KP; kk += 32) {
    v16h a, b;
    if constexpr ((K & 31) == 0) {       // fully in-bounds: 4x b128 loads
      const float4 f0 = *(const float4*)(Arow + kk + kbA);
      const float4 f1 = *(const float4*)(Arow + kk + kbA + 4);
      const float4 f2 = *(const float4*)(Arow + kk + kbA + 16);
      const float4 f3 = *(const float4*)(Arow + kk + kbA + 20);
      a[0] = (half_t)f0.x; a[1] = (half_t)f0.y; a[2]  = (half_t)f0.z; a[3]  = (half_t)f0.w;
      a[4] = (half_t)f1.x; a[5] = (half_t)f1.y; a[6]  = (half_t)f1.z; a[7]  = (half_t)f1.w;
      a[8] = (half_t)f2.x; a[9] = (half_t)f2.y; a[10] = (half_t)f2.z; a[11] = (half_t)f2.w;
      a[12] = (half_t)f3.x; a[13] = (half_t)f3.y; a[14] = (half_t)f3.z; a[15] = (half_t)f3.w;
      __builtin_prefetch(Arow + kk + 64, 0, 1);          // global_prefetch_b8
    } else {                             // K=15 zero-pad path (single step)
#pragma unroll
      for (int e = 0; e < 8; ++e) {
        int k0 = kk + kbA + e;
        int k1 = kk + kbA + 16 + e;
        a[e]     = (k0 < K) ? (half_t)Arow[k0] : (half_t)0.f;
        a[8 + e] = (k1 < K) ? (half_t)Arow[k1] : (half_t)0.f;
      }
    }
    v8h b0 = *(const v8h*)(wcol + kk);                   // ds_load_b128
    v8h b1 = *(const v8h*)(wcol + kk + 16);
#pragma unroll
    for (int e = 0; e < 8; ++e) { b[e] = b0[e]; b[8 + e] = b1[e]; }
    acc = __builtin_amdgcn_wmma_f32_16x16x32_f16(false, a, false, b,
                                                 (short)0, acc, false, false);
  }

  if (colOk) {
    float bv = 0.f;
    if constexpr (FLAGS & GF_BIAS) bv = bias[col];       // one load per lane
#pragma unroll
    for (int r = 0; r < 8; ++r) {
      float v = acc[r] + bv;
      if constexpr (FLAGS & GF_RELU) v = fmaxf(v, 0.f);
      Ccol[(long)r * N] = v;
    }
  }
}

// ------------------------------------------------ per-graph kNN (top-16 by L2)
template <int F>
__global__ __launch_bounds__(256) void knn_kernel(
    const float* __restrict__ x, int* __restrict__ knn)
{
  extern __shared__ float xs[];          // NPTS*F + NPTS floats
  const int g = blockIdx.x;
  float* sq = xs + NPTS * F;
  const float* xg = x + (long)g * NPTS * F;
  for (int t = threadIdx.x; t < NPTS * F; t += 256) xs[t] = xg[t];
  __syncthreads();
  for (int t = threadIdx.x; t < NPTS; t += 256) {
    float s = 0.f;
    const float4* r = (const float4*)(xs + t * F);
#pragma unroll
    for (int q = 0; q < (F >> 2); ++q) {
      float4 v = r[q];
      s += v.x * v.x + v.y * v.y + v.z * v.z + v.w * v.w;
    }
    sq[t] = s;
  }
  __syncthreads();

  const int i = blockIdx.y * 256 + threadIdx.x;
  float bd[KNN]; int bi[KNN];
#pragma unroll
  for (int t = 0; t < KNN; ++t) { bd[t] = 3.4e38f; bi[t] = 0; }
  const float si = sq[i];
  const float4* xi4 = (const float4*)(xs + i * F);
  for (int j = 0; j < NPTS; ++j) {
    const float4* xj4 = (const float4*)(xs + j * F);
    float dot = 0.f;
#pragma unroll
    for (int q = 0; q < (F >> 2); ++q) {
      float4 a = xi4[q], b = xj4[q];
      dot += a.x * b.x + a.y * b.y + a.z * b.z + a.w * b.w;
    }
    float d = si + sq[j] - 2.f * dot;
    if (d < bd[KNN - 1]) {               // insertion keeps earliest index on ties
      bd[KNN - 1] = d; bi[KNN - 1] = j;
#pragma unroll
      for (int t = KNN - 1; t > 0; --t) {
        if (bd[t] < bd[t - 1]) {
          float td = bd[t]; bd[t] = bd[t - 1]; bd[t - 1] = td;
          int   ti = bi[t]; bi[t] = bi[t - 1]; bi[t - 1] = ti;
        }
      }
    }
  }
  int* outp = knn + ((long)g * NPTS + i) * KNN;
#pragma unroll
  for (int t = 0; t < KNN; ++t) outp[t] = g * NPTS + bi[t];
}

// ------------------------- fused DynamicEdgeConv: msg build + WMMA MLP + max
// 16 nodes x 16 neighbors per block; messages + weights in skewed f16 LDS.
template <int F, int Fout>
__global__ __launch_bounds__(256) void edgeconv_wmma(
    const float* __restrict__ x, const int* __restrict__ knn,
    const float* __restrict__ W, const float* __restrict__ bias,
    float* __restrict__ out)
{
  constexpr int K2 = 2 * F;
  constexpr int MS = K2 + 8;             // skewed msg row stride (halves)
  constexpr int WS = K2 + 8;             // skewed W col stride (halves)
  extern __shared__ half_t sh[];
  half_t* msg = sh;                      // 256 * MS
  half_t* Wl  = sh + 256 * MS;           // Fout * WS, col-major
  const int tid = threadIdx.x;
  {
    const int ln = tid >> 4, jn = tid & 15;
    const long i = (long)blockIdx.x * 16 + ln;
    const long j = knn[i * KNN + jn];
    half_t* m = msg + tid * MS;
    const float4* xi4 = (const float4*)(x + i * F);
    const float4* xj4 = (const float4*)(x + j * F);
#pragma unroll
    for (int q = 0; q < F / 4; ++q) {
      float4 a = xi4[q], b = xj4[q];
      m[4 * q + 0] = (half_t)a.x;  m[4 * q + 1] = (half_t)a.y;
      m[4 * q + 2] = (half_t)a.z;  m[4 * q + 3] = (half_t)a.w;
      m[F + 4 * q + 0] = (half_t)(b.x - a.x);  m[F + 4 * q + 1] = (half_t)(b.y - a.y);
      m[F + 4 * q + 2] = (half_t)(b.z - a.z);  m[F + 4 * q + 3] = (half_t)(b.w - a.w);
    }
  }
  for (int t = tid; t < Fout * K2; t += 256) {
    int n = t / K2, k = t - n * K2;
    Wl[n * WS + k] = (half_t)W[k * Fout + n];
  }
  __syncthreads();

  const int wave = tid >> 5, lane = tid & 31;
  const int l15 = lane & 15, hi = lane >> 4;
  constexpr int tilesN = Fout >> 4;
  constexpr int total  = 16 * tilesN;
  for (int t = wave; t < total; t += 8) {          // uniform per wave
    const int ln = t / tilesN;
    const int tn = t - ln * tilesN;
    v8f acc;
#pragma unroll
    for (int r = 0; r < 8; ++r) acc[r] = 0.f;
    const half_t* mrow = msg + (ln * 16 + l15) * MS + (hi ? 8 : 0);
    const half_t* wcol = Wl + (tn * 16 + l15) * WS + (hi ? 8 : 0);
#pragma unroll
    for (int kk = 0; kk < K2; kk += 32) {
      v16h a, b;
      v8h a0 = *(const v8h*)(mrow + kk);           // ds_load_b128 x4
      v8h a1 = *(const v8h*)(mrow + kk + 16);
      v8h b0 = *(const v8h*)(wcol + kk);
      v8h b1 = *(const v8h*)(wcol + kk + 16);
#pragma unroll
      for (int e = 0; e < 8; ++e) {
        a[e] = a0[e]; a[8 + e] = a1[e];
        b[e] = b0[e]; b[8 + e] = b1[e];
      }
      acc = __builtin_amdgcn_wmma_f32_16x16x32_f16(false, a, false, b,
                                                   (short)0, acc, false, false);
    }
    float m = acc[0];
#pragma unroll
    for (int r = 1; r < 8; ++r) m = fmaxf(m, acc[r]);
    m = fmaxf(m, __shfl_xor(m, 16, 32));           // fold the two lane halves
    if (hi == 0) {
      const long node = (long)blockIdx.x * 16 + ln;
      out[node * Fout + tn * 16 + l15] = m + bias[tn * 16 + l15];
    }
  }
}

// --------------------------------------------------------------- batch norm
__global__ __launch_bounds__(256) void bn_kernel(
    float* __restrict__ x, const float* __restrict__ gam,
    const float* __restrict__ bet, int M, int F, int relu)
{
  __shared__ float rs[256], rq[256];
  const int f = blockIdx.x;
  float s = 0.f, q = 0.f;
  for (int r = threadIdx.x; r < M; r += 256) {
    float v = x[(long)r * F + f];
    s += v; q += v * v;
  }
  rs[threadIdx.x] = s; rq[threadIdx.x] = q;
  __syncthreads();
  for (int off = 128; off > 0; off >>= 1) {
    if (threadIdx.x < off) {
      rs[threadIdx.x] += rs[threadIdx.x + off];
      rq[threadIdx.x] += rq[threadIdx.x + off];
    }
    __syncthreads();
  }
  const float mean = rs[0] / (float)M;
  const float var  = rq[0] / (float)M - mean * mean;
  const float sc = rsqrtf(var + 1e-5f) * gam[f];
  const float sh = bet[f] - mean * sc;
  for (int r = threadIdx.x; r < M; r += 256) {
    float v = x[(long)r * F + f] * sc + sh;
    if (relu) v = fmaxf(v, 0.f);
    x[(long)r * F + f] = v;
  }
}

// ------------------------------------------------------------------ pooling
__global__ void pool_kernel(const float* __restrict__ x, float* __restrict__ pooled,
                            int F, int colOff) {
  const int g = blockIdx.x;
  const int f = threadIdx.x;
  if (f >= F) return;
  float m = -3.4e38f;
  const float* p = x + (long)g * NPTS * F + f;
  for (int r = 0; r < NPTS; ++r) m = fmaxf(m, p[(long)r * F]);
  pooled[g * 192 + colOff + f] = m;
}

__global__ void logsoftmax_kernel(const float* __restrict__ logits, float* __restrict__ out) {
  int r = threadIdx.x;
  if (r >= NB) return;
  float l0 = logits[r * 4], l1 = logits[r * 4 + 1], l2 = logits[r * 4 + 2], l3 = logits[r * 4 + 3];
  float m = fmaxf(fmaxf(l0, l1), fmaxf(l2, l3));
  float s = expf(l0 - m) + expf(l1 - m) + expf(l2 - m) + expf(l3 - m);
  float lse = m + logf(s);
  out[r * 4] = l0 - lse; out[r * 4 + 1] = l1 - lse;
  out[r * 4 + 2] = l2 - lse; out[r * 4 + 3] = l3 - lse;
}

// =========================================================== host orchestration
static inline void launch_zero(float* p, long n, hipStream_t s) {
  int blocks = (int)((n + 255) / 256);
  if (blocks > 4096) blocks = 4096;
  zero_f32<<<blocks, 256, 0, s>>>(p, n);
}

template <int K, int N>
static inline void launch_gemm(const float* A, const float* W, const float* b, float* C,
                               int M, int flags, hipStream_t s) {
  constexpr int KP  = (K + 31) & ~31;
  constexpr int NPD = (N + 15) & ~15;
  size_t lds = (size_t)NPD * (KP + 8) * sizeof(half_t);
  int tiles  = (M >> 4) * (NPD >> 4);
  int blocks = (tiles + 3) / 4;
  switch (flags) {
    case 0:
      gemm_wmma<K, N, 0><<<blocks, 128, lds, s>>>(A, W, b, C, M); break;
    case GF_BIAS:
      gemm_wmma<K, N, GF_BIAS><<<blocks, 128, lds, s>>>(A, W, b, C, M); break;
    case GF_BIAS | GF_RELU:
      gemm_wmma<K, N, GF_BIAS | GF_RELU><<<blocks, 128, lds, s>>>(A, W, b, C, M); break;
    case GF_ACCUM:
      gemm_wmma<K, N, GF_ACCUM><<<blocks, 128, lds, s>>>(A, W, b, C, M); break;
    default:
      gemm_wmma<K, N, GF_ACCUM | GF_BIAS | GF_RELU><<<blocks, 128, lds, s>>>(A, W, b, C, M); break;
  }
}

template <int KIN, int NOUT>
static void tagconv(const float* X, const float* W, const float* bias, float* outbuf,
                    int khops, float* h0, float* h1,
                    const int* src, const int* dst, const float* nrm, hipStream_t s) {
  launch_gemm<KIN, NOUT>(X, W, bias, outbuf, NNODES,
                         (khops == 1) ? (GF_BIAS | GF_RELU) : 0, s);
  const float* cur = X;
  float* bufs[2] = {h0, h1};
  for (int k = 1; k < khops; ++k) {
    float* tgt = bufs[(k - 1) & 1];
    launch_zero(tgt, (long)NNODES * KIN, s);
    hop_kernel<<<dim3((NE + 255) / 256, KIN), 256, 0, s>>>(cur, src, dst, nrm, tgt, KIN);
    int fl = GF_ACCUM | ((k == khops - 1) ? (GF_BIAS | GF_RELU) : 0);
    launch_gemm<KIN, NOUT>(tgt, W + (long)k * KIN * NOUT, bias, outbuf, NNODES, fl, s);
    cur = tgt;
  }
}

template <int F, int Fout>
static void edgeconv(const float* xin, const float* W, const float* b,
                     float* xout, int* knnb, hipStream_t s) {
  size_t knnLds = (size_t)NPTS * F * sizeof(float) + NPTS * sizeof(float);
  knn_kernel<F><<<dim3(NB, NPTS / 256), 256, knnLds, s>>>(xin, knnb);
  constexpr int K2 = 2 * F;
  size_t ecLds = ((size_t)256 * (K2 + 8) + (size_t)Fout * (K2 + 8)) * sizeof(half_t);
  edgeconv_wmma<F, Fout><<<NNODES / 16, 256, ecLds, s>>>(xin, knnb, W, b, xout);
}

extern "C" void kernel_launch(void* const* d_in, const int* in_sizes, int n_in,
                              void* d_out, int out_size, void* d_ws, size_t ws_size,
                              hipStream_t stream) {
  (void)in_sizes; (void)n_in; (void)out_size; (void)ws_size;
  const float* x    = (const float*)d_in[0];
  const int*   ei   = (const int*)d_in[1];
  const int*   src  = ei;
  const int*   dst  = ei + NE;
  const float* t11W = (const float*)d_in[3];  const float* t11b = (const float*)d_in[4];
  const float* t12W = (const float*)d_in[5];  const float* t12b = (const float*)d_in[6];
  const float* bn1g = (const float*)d_in[7];  const float* bn1b = (const float*)d_in[8];
  const float* t21W = (const float*)d_in[9];  const float* t21b = (const float*)d_in[10];
  const float* t22W = (const float*)d_in[11]; const float* t22b = (const float*)d_in[12];
  const float* bn2g = (const float*)d_in[13]; const float* bn2b = (const float*)d_in[14];
  const float* e31W = (const float*)d_in[15]; const float* e31b = (const float*)d_in[16];
  const float* e32W = (const float*)d_in[17]; const float* e32b = (const float*)d_in[18];
  const float* bn3g = (const float*)d_in[19]; const float* bn3b = (const float*)d_in[20];
  const float* e41W = (const float*)d_in[21]; const float* e41b = (const float*)d_in[22];
  const float* e42W = (const float*)d_in[23]; const float* e42b = (const float*)d_in[24];
  const float* bn4g = (const float*)d_in[25]; const float* bn4b = (const float*)d_in[26];
  const float* l1W  = (const float*)d_in[27]; const float* l1b  = (const float*)d_in[28];
  const float* m1W  = (const float*)d_in[29]; const float* m1b  = (const float*)d_in[30];
  const float* bn5g = (const float*)d_in[31]; const float* bn5b = (const float*)d_in[32];
  const float* m2W  = (const float*)d_in[33]; const float* m2b  = (const float*)d_in[34];

  size_t off = 0;
  auto carve = [&](size_t bytes) -> char* {
    char* p = (char*)d_ws + off;
    off += (bytes + 255) & ~(size_t)255;
    return p;
  };
  float* deg  = (float*)carve((size_t)NNODES * 4);
  float* nrm  = (float*)carve((size_t)NE * 4);
  float* hb0  = (float*)carve((size_t)NNODES * 64 * 4);
  float* hb1  = (float*)carve((size_t)NNODES * 64 * 4);
  float* bufA = (float*)carve((size_t)NNODES * 64 * 4);  // t11 out
  float* bufB = (float*)carve((size_t)NNODES * 64 * 4);  // t12 out -> x1
  float* bufC = (float*)carve((size_t)NNODES * 64 * 4);  // t21 out
  float* bufD = (float*)carve((size_t)NNODES * 32 * 4);  // t22 out -> x2
  float* bufE = (float*)carve((size_t)NNODES * 64 * 4);  // e31 out
  float* bufF = (float*)carve((size_t)NNODES * 64 * 4);  // e32 out -> x3
  float* bufG = (float*)carve((size_t)NNODES * 32 * 4);  // e41 out
  float* bufH = (float*)carve((size_t)NNODES * 32 * 4);  // e42 out -> x4
  int*   knnb = (int*)  carve((size_t)NNODES * KNN * 4);
  float* pooled = (float*)carve((size_t)NB * 192 * 4);
  float* p1     = (float*)carve((size_t)NB * 128 * 4);
  float* p2     = (float*)carve((size_t)NB * 32 * 4);
  float* logits = (float*)carve((size_t)NB * 4 * 4);

  // degree + symmetric norm
  launch_zero(deg, NNODES, stream);
  deg_kernel<<<(NE + 255) / 256, 256, 0, stream>>>(dst, deg, NE);
  norm_kernel<<<(NE + 255) / 256, 256, 0, stream>>>(src, dst, deg, nrm, NE);

  // stage 1: TAGConv(15->64,K=5) -> TAGConv(64->64,K=5) -> BN -> pool
  tagconv<15, 64>(x,    t11W, t11b, bufA, 5, hb0, hb1, src, dst, nrm, stream);
  tagconv<64, 64>(bufA, t12W, t12b, bufB, 5, hb0, hb1, src, dst, nrm, stream);
  bn_kernel<<<64, 256, 0, stream>>>(bufB, bn1g, bn1b, NNODES, 64, 0);
  pool_kernel<<<NB, 64, 0, stream>>>(bufB, pooled, 64, 0);

  // stage 2: TAGConv(64->64,K=4) -> TAGConv(64->32,K=4) -> BN -> pool
  tagconv<64, 64>(bufB, t21W, t21b, bufC, 4, hb0, hb1, src, dst, nrm, stream);
  tagconv<64, 32>(bufC, t22W, t22b, bufD, 4, hb0, hb1, src, dst, nrm, stream);
  bn_kernel<<<32, 256, 0, stream>>>(bufD, bn2g, bn2b, NNODES, 32, 0);
  pool_kernel<<<NB, 64, 0, stream>>>(bufD, pooled, 32, 64);

  // stage 3: EdgeConv(32->64) -> EdgeConv(64->64) -> BN -> pool
  edgeconv<32, 64>(bufD, e31W, e31b, bufE, knnb, stream);
  edgeconv<64, 64>(bufE, e32W, e32b, bufF, knnb, stream);
  bn_kernel<<<64, 256, 0, stream>>>(bufF, bn3g, bn3b, NNODES, 64, 0);
  pool_kernel<<<NB, 64, 0, stream>>>(bufF, pooled, 64, 96);

  // stage 4: EdgeConv(64->32) -> EdgeConv(32->32) -> BN -> pool
  edgeconv<64, 32>(bufF, e41W, e41b, bufG, knnb, stream);
  edgeconv<32, 32>(bufG, e42W, e42b, bufH, knnb, stream);
  bn_kernel<<<32, 256, 0, stream>>>(bufH, bn4g, bn4b, NNODES, 32, 0);
  pool_kernel<<<NB, 64, 0, stream>>>(bufH, pooled, 32, 160);

  // head: lin1 -> mlp1 -> BN+ReLU -> mlp2 -> log_softmax
  launch_gemm<192, 128>(pooled, l1W, l1b, p1, NB, GF_BIAS, stream);
  launch_gemm<128, 32>(p1, m1W, m1b, p2, NB, GF_BIAS, stream);
  bn_kernel<<<32, 256, 0, stream>>>(p2, bn5g, bn5b, NB, 32, 1);
  launch_gemm<32, 4>(p2, m2W, m2b, logits, NB, GF_BIAS, stream);
  logsoftmax_kernel<<<1, 64, 0, stream>>>(logits, (float*)d_out);
}
